// CLAPP_RSNN_6459630813607
// MI455X (gfx1250) — compile-verified, matrix operand order checked by reading, not run on hardware
//
#include <hip/hip_runtime.h>

// ---------------- problem constants ----------------
#define TT   100
#define BB   256
#define II   700
#define HH   512
#define MROWS (TT*BB)            // 25600
#define BH  131072               // B*H
#define BI  179200               // B*I

// d_out flat offsets (elements, fp32), reference return order:
// spks1[T,B,512], spk_traces[2,B,512], losses[2], dW0[B,512,700], dW1[B,512,512]
#define O_SPKS1  0L
#define O_TRACE  13107200L
#define O_LOSS   13369344L
#define O_DW0    13369346L
#define O_DW1    105119746L

// workspace byte offsets
#define WS_CUR    0L            // f32 [T,B,512]  (cur in -> g in place; reused for layer1)
#define WS_SPK0   52428800L     // bf16 [T,B,512]
#define WS_GHAT0  78643200L     // bf16 [T,B,512]
#define WS_GHAT1  104857600L    // bf16 [T,B,512]
#define WS_PMEAN  131072000L    // f32 [2*256]

typedef __attribute__((ext_vector_type(16))) __bf16 v16bf;
typedef __attribute__((ext_vector_type(8)))  float  v8f;

// ---------------- init: per-sample means of prev traces + zero loss slots ----
__global__ void init_means(const float* __restrict__ prev0,
                           const float* __restrict__ prev1,
                           float* __restrict__ pmean, float* __restrict__ loss_out) {
    int i = threadIdx.x;
    if (i < 2) loss_out[i] = 0.0f;
    if (i < 512) {
        const float* p = (i < 256) ? (prev0 + (long)i * HH)
                                   : (prev1 + (long)(i - 256) * HH);
        float s = 0.f;
        for (int h = 0; h < HH; ++h) s += p[h];
        pmean[i] = s * (1.0f / HH);
    }
}

// ---------------- fragment loads -------------------------------------------
// 16-bit A/B-matrix 16x32 VGPR layout:
//   lanes 0-15 : elems 0-7 -> K=k0+0..7,  elems 8-15 -> K=k0+16..23
//   lanes 16-31: elems 0-7 -> K=k0+8..15, elems 8-15 -> K=k0+24..31
// kbase = k0 + 8*hw; element e maps to k = kbase + (e<8 ? e : e+8).
// Full chunk: no bounds logic at all (16 plain loads, one wait, compile-time
// stride -> immediate offsets).
template <typename T, int SK>
__device__ __forceinline__ v16bf load_frag_full(const T* __restrict__ p, int kbase) {
    v16bf f;
#pragma unroll
    for (int e = 0; e < 16; ++e) {
        int kl = (e < 8) ? e : (e + 8);
        f[e] = (__bf16)(float)p[(kbase + kl) * SK];
    }
    return f;
}

// Tail chunk: clamp index, zero value with a cndmask (no exec divergence).
template <typename T, int SK>
__device__ __forceinline__ v16bf load_frag_tail(const T* __restrict__ p, int kbase, int K) {
    v16bf f;
#pragma unroll
    for (int e = 0; e < 16; ++e) {
        int kl = (e < 8) ? e : (e + 8);
        int k  = kbase + kl;
        int kc = (k < K) ? k : 0;
        float v = (float)p[kc * SK];
        f[e] = (__bf16)((k < K) ? v : 0.0f);
    }
    return f;
}

// ---------------- fully-static batched WMMA GEMM ---------------------------
// C[b][m,n] = sum_k A[b][m,k]*B[b][n,k]  (bf16 multiply, f32 accumulate)
// One wave computes a 16 x (16*NT) strip; A fragment reused across NT WMMAs.
// All strides/dims are template constants -> 32-bit address math, no spills.
// grid = (M/16, ceil(N/16)/NT, batch), block = 32.
template <typename TA, typename TB, int NT,
          int SAM, int SAK, int SBN, int SBK,
          int M, int N, int K, int LDC>
__global__ void wmma_gemm(const TA* __restrict__ A, int sAb,
                          const TB* __restrict__ Bm, int sBb,
                          float* __restrict__ C, int sCb) {
    static_assert(M % 16 == 0, "M must be tile-aligned");
    const int lane = threadIdx.x & 31;
    const int hw   = lane >> 4;
    const int lmod = lane & 15;
    const int mt   = blockIdx.x;
    const int nt0  = blockIdx.y * NT;
    A  += (size_t)blockIdx.z * (size_t)sAb;
    Bm += (size_t)blockIdx.z * (size_t)sBb;
    C  += (size_t)blockIdx.z * (size_t)sCb;

    const int mrow = mt * 16 + lmod;                 // always < M
    const TA* __restrict__ pa = A + (size_t)mrow * SAM;
    const TB* __restrict__ pb[NT];
#pragma unroll
    for (int j = 0; j < NT; ++j) {
        int ncol = (nt0 + j) * 16 + lmod;
        if (ncol > N - 1) ncol = N - 1;   // clamp: garbage columns are never stored
        pb[j] = Bm + (size_t)ncol * SBN;
    }

    v8f acc[NT];
#pragma unroll
    for (int j = 0; j < NT; ++j) acc[j] = (v8f){};

    constexpr int KF = (K / 32) * 32;
    const int kb0 = 8 * hw;
    for (int k0 = 0; k0 < KF; k0 += 32) {
        v16bf af = load_frag_full<TA, SAK>(pa, k0 + kb0);
#pragma unroll
        for (int j = 0; j < NT; ++j) {
            v16bf bf = load_frag_full<TB, SBK>(pb[j], k0 + kb0);
            acc[j] = __builtin_amdgcn_wmma_f32_16x16x32_bf16(
                false, af, false, bf, (short)0, acc[j], false, false);
        }
    }
    if constexpr (KF < K) {   // statically-known ragged tail (K=700 or K=100)
        v16bf af = load_frag_tail<TA, SAK>(pa, KF + kb0, K);
#pragma unroll
        for (int j = 0; j < NT; ++j) {
            v16bf bf = load_frag_tail<TB, SBK>(pb[j], KF + kb0, K);
            acc[j] = __builtin_amdgcn_wmma_f32_16x16x32_bf16(
                false, af, false, bf, (short)0, acc[j], false, false);
        }
    }

#pragma unroll
    for (int j = 0; j < NT; ++j) {
#pragma unroll
        for (int r = 0; r < 8; ++r) {   // C layout: VGPR r -> M=r+8*hw, N=lane&15
            int m = mt * 16 + r + 8 * hw;
            int n = (nt0 + j) * 16 + lmod;
            if (n < N) C[(size_t)m * LDC + n] = acc[j][r];
        }
    }
}

// ---------------- LIF scan (forward over T), one thread per (b,h) ----------
// Reads cur, overwrites it in place with g = prev * surrogate(mem-1).
__global__ void lif_scan(const float* __restrict__ prev, const float* __restrict__ pmean,
                         float* __restrict__ curg,
                         __bf16* __restrict__ spk_bf,   // layer0 spikes (or null)
                         float*  __restrict__ spk_f,    // layer1 spikes -> d_out (or null)
                         float*  __restrict__ trace_out,
                         float*  __restrict__ loss_out) {
    const int idx = blockIdx.x * blockDim.x + threadIdx.x;  // b*H + h
    if (idx >= BH) return;
    const int b = idx >> 9;                                  // H == 512
    const float pv = prev[idx];
    const float fb = pv - pmean[b];
    const float beta = 0.95f, pi = 3.14159265358979f;
    float mem = 0.f, tr = 0.f, lsum = 0.f;
    for (int t = 0; t < TT; ++t) {
        int off = t * BH + idx;
        if (t + 1 < TT) __builtin_prefetch(curg + off + BH, 0, 0);
        float cur   = curg[off];
        float reset = (mem > 1.0f) ? 1.0f : 0.0f;   // from previous mem
        mem = beta * mem + cur - reset;
        float spk = (mem > 1.0f) ? 1.0f : 0.0f;
        tr   += spk * (1.0f / TT);
        lsum += spk * fb;
        if (spk_bf) spk_bf[off] = (__bf16)spk;
        if (spk_f)  spk_f[off]  = spk;
        float x = pi * (mem - 1.0f);
        curg[off] = pv * (1.0f / (pi * (1.0f + x * x)));   // g
    }
    trace_out[idx] = tr;
    atomicAdd(loss_out, -lsum * (1.0f / BB));   // losses[l] = mean_b( sum_t -spk.fb )
}

// ---------------- reverse exponential scan: ghat[t] = g[t] + beta*ghat[t+1] -
__global__ void rev_scan(const float* __restrict__ g, __bf16* __restrict__ ghat) {
    const int idx = blockIdx.x * blockDim.x + threadIdx.x;
    if (idx >= BH) return;
    float acc = 0.f;
    for (int t = TT - 1; t >= 0; --t) {
        int off = t * BH + idx;
        acc = g[off] + 0.95f * acc;
        ghat[off] = (__bf16)acc;
    }
}

// ---------------------------------------------------------------------------
extern "C" void kernel_launch(void* const* d_in, const int* in_sizes, int n_in,
                              void* d_out, int out_size, void* d_ws, size_t ws_size,
                              hipStream_t stream) {
    (void)in_sizes; (void)n_in; (void)out_size; (void)ws_size;
    const float* inp   = (const float*)d_in[0];   // [T,B,700]
    const float* W0    = (const float*)d_in[1];   // [512,700]
    const float* W1    = (const float*)d_in[2];   // [512,512]
    const float* prev0 = (const float*)d_in[3];   // [B,512]
    const float* prev1 = (const float*)d_in[4];   // [B,512]

    float*  out    = (float*)d_out;
    char*   ws     = (char*)d_ws;
    float*  curbuf = (float*) (ws + WS_CUR);
    __bf16* spk0   = (__bf16*)(ws + WS_SPK0);
    __bf16* ghat0  = (__bf16*)(ws + WS_GHAT0);
    __bf16* ghat1  = (__bf16*)(ws + WS_GHAT1);
    float*  pmean  = (float*) (ws + WS_PMEAN);
    float*  losses = out + O_LOSS;

    // 0) per-sample means of prev traces; zero loss accumulators
    init_means<<<1, 512, 0, stream>>>(prev0, prev1, pmean, losses);

    // 1) cur0 = inp @ W0^T : [25600,700] x [512,700]^T
    wmma_gemm<float, float, 4, II, 1, II, 1, MROWS, HH, II, HH>
        <<<dim3(MROWS / 16, HH / 64, 1), 32, 0, stream>>>(inp, 0, W0, 0, curbuf, 0);

    // 2) LIF scan layer 0: spikes (bf16), trace0, loss0, g0 in place
    lif_scan<<<BH / 256, 256, 0, stream>>>(
        prev0, pmean, curbuf, spk0, nullptr, out + O_TRACE, losses + 0);

    // 3) ghat0[t] = g0[t] + beta*ghat0[t+1]
    rev_scan<<<BH / 256, 256, 0, stream>>>(curbuf, ghat0);

    // 4) cur1 = spk0 @ W1^T (reuse curbuf)
    wmma_gemm<__bf16, float, 4, HH, 1, HH, 1, MROWS, HH, HH, HH>
        <<<dim3(MROWS / 16, HH / 64, 1), 32, 0, stream>>>(spk0, 0, W1, 0, curbuf, 0);

    // 5) LIF scan layer 1: spks1 (f32, straight to d_out), trace1, loss1, g1 in place
    lif_scan<<<BH / 256, 256, 0, stream>>>(
        prev1, pmean + 256, curbuf, nullptr, out + O_SPKS1,
        out + O_TRACE + BH, losses + 1);

    // 6) ghat1
    rev_scan<<<BH / 256, 256, 0, stream>>>(curbuf, ghat1);

    // 7) dW0[b] = Ghat0_b [512,100] @ Inp_b [100,700]
    //    A(m,k)=ghat0[k*BH + b*512 + m]; B(n,k)=inp[k*BI + b*700 + n]
    //    N = 700 -> 44 tiles -> 11 groups of 4 (edge tile clamped)
    wmma_gemm<__bf16, float, 4, 1, BH, 1, BI, HH, II, TT, II>
        <<<dim3(HH / 16, 11, BB), 32, 0, stream>>>(
            ghat0, HH, inp, II, out + O_DW0, HH * II);

    // 8) dW1[b] = Ghat1_b [512,100] @ Spk0_b [100,512]
    wmma_gemm<__bf16, __bf16, 4, 1, BH, 1, BH, HH, HH, TT, HH>
        <<<dim3(HH / 16, HH / 64, BB), 32, 0, stream>>>(
            ghat1, HH, spk0, HH, out + O_DW1, HH * HH);
}
